// LstmEncoder_71038759076377
// MI455X (gfx1250) — matrix operand order repaired and checked by reading, not compiled, not run on hardware
//
#include <hip/hip_runtime.h>
#include <hip/hip_bf16.h>
#include <stdint.h>

// ---------------------------------------------------------------------------
// Stacked 2-layer LSTM (B=64, T=1024, D=256, H=1024) for gfx1250.
// bf16 WMMA (16x16x32, f32 accum) for the per-step fused GEMM
//   g = [x_t | h_prev] @ [W_ih | W_hh]^T + (b_ih + b_hh)
// Weights packed once to bf16 [4H, K]; 26 MB total -> resident in 192 MB L2
// across the whole scan. Recurrence = per-timestep launches (stream order is
// the dependency), ping-pong h/c buffers avoid cross-workgroup races.
// This revision: wave-uniform (readfirstlane) gate/k-half indices + split
// x-region / h-region loops so the hot loop is scalar-controlled with no
// per-iteration pointer select.
// ---------------------------------------------------------------------------

#define LSTM_B 64
#define LSTM_T 1024
#define LSTM_D 256
#define LSTM_H 1024

typedef __attribute__((ext_vector_type(16))) __bf16 bf16x16;
typedef __attribute__((ext_vector_type(8)))  float  f32x8;

union FragB16 {
  bf16x16 v;
  uint4 q[2];
};

__device__ __forceinline__ float sigmoid_f(float x) {
  return 1.0f / (1.0f + __expf(-x));
}
__device__ __forceinline__ float tanh_f(float x) {
  float e = __expf(-2.0f * fabsf(x));
  float t = (1.0f - e) / (1.0f + e);
  return copysignf(t, x);
}

// One LSTM timestep for one layer.
//   X      : bf16, logical [B x Kx] with row stride ldx (elements)
//   Hprev  : bf16, [B x H]
//   W      : bf16, [4H x K] row-major, row = gate*H + n, cols = [x-part | h-part]
//   bias   : f32,  [4H] (b_ih + b_hh)
//   Cin/Cout : f32 [B x H],  Hout : bf16 [B x H]
// Block: 256 threads = 8 waves = 4 gates x 2 K-halves. Grid: (B/16, H/16).
__global__ __launch_bounds__(256)
void lstm_step_kernel(const __bf16* __restrict__ X, long long ldx, int Kx,
                      const __bf16* __restrict__ Hprev,
                      const __bf16* __restrict__ W, int K,
                      const float*  __restrict__ bias,
                      const float*  __restrict__ Cin,
                      float*        __restrict__ Cout,
                      __bf16*       __restrict__ Hout)
{
  __shared__ float gbuf[2][4][16][16];   // [k-half][gate][m][n]
  const int H    = LSTM_H;
  const int tid  = threadIdx.x;
  // wave id is uniform across the wave: pin it to an SGPR so the K loops are
  // compiled as uniform (scalar branch) loops instead of EXEC-masked ones.
  const int wave = __builtin_amdgcn_readfirstlane(tid >> 5);
  const int lane = tid & 31;
  const int gate = wave & 3;             // PyTorch order: i, f, g, o
  const int kh   = wave >> 2;            // which K half this wave accumulates
  const int m_base = blockIdx.x * 16;    // batch tile
  const int n_base = blockIdx.y * 16;    // hidden tile

  const int hiHalf = (lane >= 16) ? 1 : 0;
  const int aoff   = hiHalf ? 8 : 0;     // A: lanes 16-31 take K+8 sub-chunk
  const int am     = lane & 15;          // A row (M) within tile
  const int bn     = lane & 15;          // B column (N) within tile
  const int boff   = hiHalf ? 8 : 0;     // B: lanes 16-31 take K+8 sub-chunk

  const __bf16* xrow = X     + (long long)(m_base + am) * ldx + aoff;
  const __bf16* hrow = Hprev + (size_t)(m_base + am) * H + aoff;
  const __bf16* wrow = W     + (size_t)(gate * H + n_base + bn) * K + boff;

  const int kHalf = K >> 1;              // multiple of 32 for K=1280/2048
  const int kBeg  = kh * kHalf;          // SGPR (kh uniform)
  const int kEnd  = kBeg + kHalf;        // SGPR
  const int xEnd  = (Kx < kEnd) ? Kx : kEnd;   // end of x-region for this wave

  f32x8 acc = {};
  int k0 = kBeg;

  // ---- x-region: A rows come from X ----
#pragma unroll 2
  for (; k0 < xEnd; k0 += 32) {
    FragB16 a, b;
    const __bf16* ap = xrow + k0;
    a.q[0] = *(const uint4*)(ap);          // K = k0+aoff    ..+7
    a.q[1] = *(const uint4*)(ap + 16);     // K = k0+16+aoff ..+7
    const __bf16* wp = wrow + k0;
    b.q[0] = *(const uint4*)(wp);          // K = k0+boff    ..+7
    b.q[1] = *(const uint4*)(wp + 16);     // K = k0+16+boff ..+7
    acc = __builtin_amdgcn_wmma_f32_16x16x32_bf16(false, a.v, false, b.v,
                                                  (short)0, acc, false, false);
  }

  // ---- h-region: A rows come from Hprev ----
#pragma unroll 2
  for (; k0 < kEnd; k0 += 32) {
    FragB16 a, b;
    const __bf16* ap = hrow + (k0 - Kx);
    a.q[0] = *(const uint4*)(ap);
    a.q[1] = *(const uint4*)(ap + 16);
    const __bf16* wp = wrow + k0;
    b.q[0] = *(const uint4*)(wp);
    b.q[1] = *(const uint4*)(wp + 16);
    acc = __builtin_amdgcn_wmma_f32_16x16x32_bf16(false, a.v, false, b.v,
                                                  (short)0, acc, false, false);
  }

  if (kh == 0) {
    const float bv = bias[gate * H + n_base + bn];  // bias depends on column n
#pragma unroll
    for (int r = 0; r < 8; ++r) acc[r] += bv;
  }
  const int mrow = hiHalf ? 8 : 0;       // C/D layout: VGPR r -> M = r (+8)
#pragma unroll
  for (int r = 0; r < 8; ++r) gbuf[kh][gate][mrow + r][bn] = acc[r];
  __syncthreads();

  // Fused gate update: 256 threads cover the 16x16 tile.
  {
    const int m = tid >> 4;
    const int n = tid & 15;
    const float gi = gbuf[0][0][m][n] + gbuf[1][0][m][n];
    const float gf = gbuf[0][1][m][n] + gbuf[1][1][m][n];
    const float gg = gbuf[0][2][m][n] + gbuf[1][2][m][n];
    const float go = gbuf[0][3][m][n] + gbuf[1][3][m][n];
    const size_t idx = (size_t)(m_base + m) * H + (n_base + n);
    const float i = sigmoid_f(gi);
    const float f = sigmoid_f(gf);
    const float g = tanh_f(gg);
    const float o = sigmoid_f(go);
    const float c = f * Cin[idx] + i * g;
    Cout[idx] = c;
    Hout[idx] = (__bf16)(o * tanh_f(c));
  }
}

__global__ void f32_to_bf16_kernel(const float* __restrict__ in,
                                   __bf16* __restrict__ out, size_t n) {
  size_t i = (size_t)blockIdx.x * blockDim.x + threadIdx.x;
  const size_t stride = (size_t)gridDim.x * blockDim.x;
  for (; i < n; i += stride) out[i] = (__bf16)in[i];
}

// Pack [W_ih | W_hh] row-major into bf16 [4H x K].
__global__ void pack_w_kernel(const float* __restrict__ Wih,
                              const float* __restrict__ Whh,
                              __bf16* __restrict__ out, int K, int Kx) {
  const int row = blockIdx.y;                          // 0 .. 4H-1
  const int k   = blockIdx.x * blockDim.x + threadIdx.x;
  if (k >= K) return;
  const float v = (k < Kx) ? Wih[(size_t)row * Kx + k]
                           : Whh[(size_t)row * LSTM_H + (k - Kx)];
  out[(size_t)row * K + k] = (__bf16)v;
}

__global__ void bias_sum_kernel(const float* __restrict__ a,
                                const float* __restrict__ b,
                                float* __restrict__ out, int n) {
  const int i = blockIdx.x * blockDim.x + threadIdx.x;
  if (i < n) out[i] = a[i] + b[i];
}

__global__ void zero_u32_kernel(uint32_t* __restrict__ p, size_t nwords) {
  size_t i = (size_t)blockIdx.x * blockDim.x + threadIdx.x;
  const size_t stride = (size_t)gridDim.x * blockDim.x;
  for (; i < nwords; i += stride) p[i] = 0u;
}

// d_out = concat(h [L,B,H] f32, c [L,B,H] f32)
__global__ void finalize_kernel(const __bf16* __restrict__ h0,
                                const __bf16* __restrict__ h1,
                                const float*  __restrict__ c0,
                                const float*  __restrict__ c1,
                                float* __restrict__ out, int BH) {
  const int i = blockIdx.x * blockDim.x + threadIdx.x;
  if (i >= BH) return;
  out[i]          = (float)h0[i];
  out[BH + i]     = (float)h1[i];
  out[2 * BH + i] = c0[i];
  out[3 * BH + i] = c1[i];
}

extern "C" void kernel_launch(void* const* d_in, const int* in_sizes, int n_in,
                              void* d_out, int out_size, void* d_ws, size_t ws_size,
                              hipStream_t stream) {
  (void)in_sizes; (void)n_in; (void)out_size; (void)ws_size;
  const int B = LSTM_B, T = LSTM_T, D = LSTM_D, H = LSTM_H;
  const int K0 = D + H;       // 1280
  const int K1 = H + H;       // 2048
  const int H4 = 4 * H;       // 4096
  const size_t BH = (size_t)B * H;

  const float* x    = (const float*)d_in[0];
  const float* Wih0 = (const float*)d_in[1];
  const float* Whh0 = (const float*)d_in[2];
  const float* bih0 = (const float*)d_in[3];
  const float* bhh0 = (const float*)d_in[4];
  const float* Wih1 = (const float*)d_in[5];
  const float* Whh1 = (const float*)d_in[6];
  const float* bih1 = (const float*)d_in[7];
  const float* bhh1 = (const float*)d_in[8];

  // ---- workspace carving (≈ 62 MB total) ----
  char* ws = (char*)d_ws;
  auto carve = [&](size_t bytes) -> char* {
    char* p = ws;
    ws += (bytes + 255) & ~(size_t)255;
    return p;
  };
  __bf16* x_bf  = (__bf16*)carve((size_t)B * T * D * 2);   // 33.5 MB
  __bf16* Wc0   = (__bf16*)carve((size_t)H4 * K0 * 2);     // 10.5 MB
  __bf16* Wc1   = (__bf16*)carve((size_t)H4 * K1 * 2);     // 16.8 MB
  float*  bias0 = (float*)carve((size_t)H4 * 4);
  float*  bias1 = (float*)carve((size_t)H4 * 4);
  char*   states = carve(4 * BH * 2 + 4 * BH * 4);         // 1.5 MB, zeroed
  __bf16* h0[2] = { (__bf16*)states,              (__bf16*)(states + BH * 2) };
  __bf16* h1[2] = { (__bf16*)(states + 2*BH*2),   (__bf16*)(states + 3*BH*2) };
  float*  c0[2] = { (float*)(states + 4*BH*2),            (float*)(states + 4*BH*2 + BH*4) };
  float*  c1[2] = { (float*)(states + 4*BH*2 + 2*BH*4),   (float*)(states + 4*BH*2 + 3*BH*4) };

  // ---- one-time prep (re-done every call: deterministic, inputs untouched) ----
  {
    const size_t nx = (size_t)B * T * D;
    f32_to_bf16_kernel<<<4096, 256, 0, stream>>>(x, x_bf, nx);
  }
  pack_w_kernel<<<dim3((K0 + 255) / 256, H4), 256, 0, stream>>>(Wih0, Whh0, Wc0, K0, D);
  pack_w_kernel<<<dim3((K1 + 255) / 256, H4), 256, 0, stream>>>(Wih1, Whh1, Wc1, K1, H);
  bias_sum_kernel<<<(H4 + 255) / 256, 256, 0, stream>>>(bih0, bhh0, bias0, H4);
  bias_sum_kernel<<<(H4 + 255) / 256, 256, 0, stream>>>(bih1, bhh1, bias1, H4);
  {
    const size_t nwords = (4 * BH * 2 + 4 * BH * 4) / 4;
    zero_u32_kernel<<<512, 256, 0, stream>>>((uint32_t*)states, nwords);
  }

  // ---- recurrent scan: stream order enforces the step dependency ----
  const dim3 sgrid(B / 16, H / 16);   // (4, 64)
  for (int t = 0; t < T; ++t) {
    const int cur = t & 1, nxt = cur ^ 1;
    // layer 0: x-part = x[:, t, :], ld = T*D
    lstm_step_kernel<<<sgrid, 256, 0, stream>>>(
        x_bf + (size_t)t * D, (long long)T * D, D,
        h0[cur], Wc0, K0, bias0, c0[cur], c0[nxt], h0[nxt]);
    // layer 1: x-part = h0_t (just produced), ld = H
    lstm_step_kernel<<<sgrid, 256, 0, stream>>>(
        h0[nxt], (long long)H, H,
        h1[cur], Wc1, K1, bias1, c1[cur], c1[nxt], h1[nxt]);
  }

  // T is even -> final states live in buffer index 0.
  finalize_kernel<<<((int)BH + 255) / 256, 256, 0, stream>>>(
      h0[0], h1[0], c0[0], c1[0], (float*)d_out, (int)BH);
}
